// ReformerClassification2_61692910240413
// MI455X (gfx1250) — compile-verified, matrix operand order checked by reading, not run on hardware
//
#include <hip/hip_runtime.h>
#include <hip/hip_bf16.h>
#include <math.h>

// ---------------- model constants ----------------
constexpr int kB = 16, kT = 512, kD = 512, kHEADS = 8, kDH = 64;
constexpr int kDEPTH = 8, kNHASH = 4, kBUCKET = 64, kNB = kT / kBUCKET; // 8
constexpr int kNCHUNK = kNHASH * kNB;                                  // 32
constexpr int kFF = 4 * kD;                                            // 2048
constexpr int kNLABELS = 2;
constexpr int kROWS = kB * kT;                                         // 8192
constexpr int kBH = kB * kHEADS;                                       // 128
constexpr int kSORTN = kNHASH * kT;                                    // 2048
constexpr float kMASK = -50000.0f;

typedef __attribute__((ext_vector_type(16))) _Float16 v16h;
typedef __attribute__((ext_vector_type(8)))  float    v8f;

// Fragment K index for 16-bit A/B WMMA layout (CDNA5 ISA):
// lane groups 0-15 / 16-31 hold K-halves; VGPR p holds K pair.
__device__ __forceinline__ int frag_k(int p, int hi) {
  return ((p & 3) << 1) + (hi << 3) + ((p >> 2) << 4);
}

// LDS offset of a __shared__ object: generic LDS addresses carry the LDS
// offset in the low 32 bits (ISA 10.2 aperture rules), which is exactly what
// the async-to-LDS VDST operand wants.
__device__ __forceinline__ unsigned lds_off(const void* p) {
  return (unsigned)(size_t)p;
}
__device__ __forceinline__ void async_ld_b128(unsigned lds, const void* g) {
  asm volatile("global_load_async_to_lds_b128 %0, %1, off"
               :: "v"(lds), "v"((unsigned long long)g) : "memory");
}
__device__ __forceinline__ void wait_async0() {
  asm volatile("s_wait_asynccnt 0" ::: "memory");
}

// ---------------- embedding ----------------
__global__ __launch_bounds__(256) void embed_kernel(
    const int* __restrict__ ids, const float* __restrict__ tok,
    const float* __restrict__ pos, float* __restrict__ x1, float* __restrict__ x2) {
  int row = blockIdx.x;              // b*T + t
  int t = row & (kT - 1);
  int id = ids[row];
  int d = threadIdx.x;
  size_t o = (size_t)row * kD;
  float a0 = tok[(size_t)id * kD + d]       + pos[t * kD + d];
  float a1 = tok[(size_t)id * kD + d + 256] + pos[t * kD + d + 256];
  x1[o + d] = a0;       x2[o + d] = a0;
  x1[o + d + 256] = a1; x2[o + d + 256] = a1;
}

// ---------------- weight cast+transpose: WT[n][k] = (f16)W[k][n] ----------------
__global__ __launch_bounds__(256) void cast_transpose_kernel(
    const float* __restrict__ W, _Float16* __restrict__ WT, int K, int N) {
  __shared__ float tile[32][33];
  int k0 = blockIdx.x * 32, n0 = blockIdx.y * 32;
  int tx = threadIdx.x & 31, ty = threadIdx.x >> 5;   // 32 x 8
  for (int r = ty; r < 32; r += 8)
    tile[r][tx] = W[(size_t)(k0 + r) * N + n0 + tx];
  __syncthreads();
  for (int r = ty; r < 32; r += 8)
    WT[(size_t)(n0 + r) * K + k0 + tx] = (_Float16)tile[tx][r];
}

// ---------------- layernorm -> f16 (one row / block, 256 thr) ----------------
__global__ __launch_bounds__(256) void layernorm_kernel(
    const float* __restrict__ x, const float* __restrict__ g,
    const float* __restrict__ bb, _Float16* __restrict__ y) {
  int row = blockIdx.x, tid = threadIdx.x;
  const float* xr = x + (size_t)row * kD;
  float v0 = xr[tid], v1 = xr[tid + 256];
  float s1 = v0 + v1, s2 = v0 * v0 + v1 * v1;
  __shared__ float r1[8], r2[8];
  __shared__ float smean, srstd;
  for (int o = 16; o > 0; o >>= 1) {
    s1 += __shfl_down(s1, o, 32);
    s2 += __shfl_down(s2, o, 32);
  }
  if ((tid & 31) == 0) { r1[tid >> 5] = s1; r2[tid >> 5] = s2; }
  __syncthreads();
  if (tid == 0) {
    float S1 = 0.f, S2 = 0.f;
    for (int i = 0; i < 8; ++i) { S1 += r1[i]; S2 += r2[i]; }
    float m = S1 * (1.0f / kD);
    float var = S2 * (1.0f / kD) - m * m;
    smean = m; srstd = rsqrtf(var + 1e-5f);
  }
  __syncthreads();
  size_t o = (size_t)row * kD;
  y[o + tid]       = (_Float16)((v0 - smean) * srstd * g[tid] + bb[tid]);
  y[o + tid + 256] = (_Float16)((v1 - smean) * srstd * g[tid + 256] + bb[tid + 256]);
}

// ---------------- fused GEMM: C = act(A @ W + bias) (+ residual) ----------------
// A: [M,K] f16, WT: [N,K] f16 (pre-transposed). Block tile 128x64, 8 waves.
// Double-buffered tile staging via async global->LDS copies (ASYNCcnt).
__global__ __launch_bounds__(256) void gemm_f16wmma(
    const _Float16* __restrict__ A, const _Float16* __restrict__ WT,
    const float* __restrict__ bias, const float* __restrict__ res,
    float* __restrict__ Cf, _Float16* __restrict__ Ch,
    int M, int N, int K, int act) {
  constexpr int TM = 128, TN = 64, TK = 32;
  __shared__ alignas(16) _Float16 sA[2][TM * TK];  // 2 x 8KB, [row][k]
  __shared__ alignas(16) _Float16 sB[2][TN * TK];  // 2 x 4KB, [n][k]
  int tid = threadIdx.x;
  int wave = tid >> 5, lane = tid & 31;
  int hi = lane >> 4, r16 = lane & 15;
  int bm = blockIdx.x * TM, bn = blockIdx.y * TN;
  int wm = (wave >> 1) * 32, wn = (wave & 1) * 32;
  v8f acc[2][2] = {};
  const int NT = K / TK;

  auto issue = [&](int kt, int buf) {
    int k0 = kt * TK;
    // A tile: 128x32 halves = 512 x 16B chunks, 2 per thread
#pragma unroll
    for (int ii = 0; ii < 2; ++ii) {
      int i = tid + ii * 256;
      int r = i >> 2, c8 = (i & 3) * 8;
      async_ld_b128(lds_off(&sA[buf][r * TK + c8]),
                    &A[(size_t)(bm + r) * K + k0 + c8]);
    }
    // B tile: 64x32 halves = 256 x 16B chunks, 1 per thread
    {
      int r = tid >> 2, c8 = (tid & 3) * 8;
      async_ld_b128(lds_off(&sB[buf][r * TK + c8]),
                    &WT[(size_t)(bn + r) * K + k0 + c8]);
    }
  };

  issue(0, 0);
  for (int kt = 0; kt < NT; ++kt) {
    wait_async0();        // this wave's tile kt landed in LDS
    __syncthreads();      // all waves' tiles landed; prev compute finished
    if (kt + 1 < NT) issue(kt + 1, (kt + 1) & 1);   // overlap with compute
    const _Float16* cA = sA[kt & 1];
    const _Float16* cB = sB[kt & 1];
    union U { v16h v; unsigned u[8]; };
    U af[2], bf[2];
#pragma unroll
    for (int tI = 0; tI < 2; ++tI) {
      int arow = wm + tI * 16 + r16;
      int bcol = wn + tI * 16 + r16;
#pragma unroll
      for (int p = 0; p < 8; ++p) {
        int k = frag_k(p, hi);
        af[tI].u[p] = *(const unsigned*)&cA[arow * TK + k];
        bf[tI].u[p] = *(const unsigned*)&cB[bcol * TK + k];
      }
    }
#pragma unroll
    for (int i = 0; i < 2; ++i)
#pragma unroll
      for (int j = 0; j < 2; ++j)
        acc[i][j] = __builtin_amdgcn_wmma_f32_16x16x32_f16(
            false, af[i].v, false, bf[j].v, (short)0, acc[i][j], false, false);
  }
  // epilogue
#pragma unroll
  for (int i = 0; i < 2; ++i) {
#pragma unroll
    for (int j = 0; j < 2; ++j) {
      int n = bn + wn + j * 16 + r16;
      float bval = bias ? bias[n] : 0.0f;
#pragma unroll
      for (int p = 0; p < 8; ++p) {
        int m = bm + wm + i * 16 + p + hi * 8;
        float x = acc[i][j][p] + bval;
        if (act == 1) x = 0.5f * x * (1.0f + erff(x * 0.70710678f)); // exact GELU
        size_t idx = (size_t)m * N + n;
        if (res) x += res[idx];
        if (Cf) Cf[idx] = x;
        if (Ch) Ch[idx] = (_Float16)x;
      }
    }
  }
}

// ---------------- LSH bucketing + bitonic argsort (one bh / block) ----------------
__global__ __launch_bounds__(256) void lsh_sort_kernel(
    const float* __restrict__ qk, const float* __restrict__ rot,
    int* __restrict__ perm) {
  __shared__ float sRot[kDH * kNHASH * (kNB / 2)];  // 64*4*4 = 1024 f32
  __shared__ unsigned keys[kSORTN];                  // 8KB
  int bh = blockIdx.x;
  int b = bh >> 3, h = bh & 7;
  int tid = threadIdx.x;
  for (int i = tid; i < kDH * kNHASH * (kNB / 2); i += 256) sRot[i] = rot[i];
  __syncthreads();
  for (int i = tid; i < kSORTN; i += 256) {
    int hh = i >> 9, t = i & (kT - 1);
    const float* q = qk + ((size_t)(b * kT + t)) * kD + h * kDH;
    float r0 = 0.f, r1 = 0.f, r2 = 0.f, r3 = 0.f;
    for (int d = 0; d < kDH; ++d) {
      float qd = q[d];
      const float* rr = &sRot[(d * kNHASH + hh) * (kNB / 2)];
      r0 += qd * rr[0]; r1 += qd * rr[1]; r2 += qd * rr[2]; r3 += qd * rr[3];
    }
    float rv[4] = {r0, r1, r2, r3};
    int bkt = 0; float best = rv[0];
    for (int j = 1; j < 4; ++j) if (rv[j] > best) { best = rv[j]; bkt = j; }
    for (int j = 0; j < 4; ++j) if (-rv[j] > best) { best = -rv[j]; bkt = 4 + j; }
    bkt += hh * kNB;
    keys[i] = ((unsigned)(bkt * kT + t) << 11) | (unsigned)i;  // unique keys
  }
  for (int ksz = 2; ksz <= kSORTN; ksz <<= 1) {
    for (int jj = ksz >> 1; jj > 0; jj >>= 1) {
      __syncthreads();
      for (int idx = tid; idx < kSORTN; idx += 256) {
        int ixj = idx ^ jj;
        if (ixj > idx) {
          unsigned a = keys[idx], c = keys[ixj];
          bool up = ((idx & ksz) == 0);
          if ((a > c) == up) { keys[idx] = c; keys[ixj] = a; }
        }
      }
    }
  }
  __syncthreads();
  for (int idx = tid; idx < kSORTN; idx += 256)
    perm[(size_t)bh * kSORTN + idx] = (int)(keys[idx] & 2047u);
}

// ---------------- chunked LSH attention (one (bh,chunk) / block) ----------------
__global__ __launch_bounds__(256) void lsh_attn_kernel(
    const float* __restrict__ qk, const float* __restrict__ vv,
    const int* __restrict__ perm, float* __restrict__ ohash,
    float* __restrict__ lseo) {
  int c = blockIdx.x;     // chunk
  int bh = blockIdx.y;
  int b = bh >> 3, h = bh & 7;
  int tid = threadIdx.x;
  int lane = tid & 31, wave = tid >> 5;
  int hi = lane >> 4, r16 = lane & 15;
  __shared__ alignas(16) _Float16 sQ[64 * 64];    //  8KB  q * DH^-0.5
  __shared__ alignas(16) _Float16 sKV[128 * 64];  // 16KB  K^T then V^T
  __shared__ float sDots[64 * 128];               // 32KB
  __shared__ int sIdx[128];
  __shared__ float sLse[64];
  const int* pRow = perm + (size_t)bh * kSORTN;

  if (tid < 64) {                       // q rows (chunk c)
    int i = pRow[c * 64 + tid];
    int t = i & (kT - 1);
    sIdx[tid] = t;
    const float* src = qk + ((size_t)(b * kT + t)) * kD + h * kDH;
    for (int e = 0; e < kDH; ++e)
      sQ[tid * 64 + e] = (_Float16)(src[e] * 0.125f);
  } else if (tid < 192) {               // k rows (chunk c + look-one-back)
    int jj = tid - 64;
    int ch = (jj < 64) ? c : ((c + kNCHUNK - 1) & (kNCHUNK - 1));
    int i = pRow[ch * 64 + (jj & 63)];
    int t = i & (kT - 1);
    if (jj >= 64) sIdx[jj] = t;
    const float* src = qk + ((size_t)(b * kT + t)) * kD + h * kDH;
    float ss = 0.f;
    for (int e = 0; e < kDH; ++e) { float x = src[e]; ss += x * x; }
    float inv = 1.0f / fmaxf(sqrtf(ss), 1e-6f);
    for (int e = 0; e < kDH; ++e)
      sKV[jj * 64 + e] = (_Float16)(src[e] * inv);
  }
  __syncthreads();

  // dots = q @ k^T : 64x128, K=64. wave w: mtile=w>>1, ntiles=(w&1)*4..+3
  {
    v8f acc[4] = {};
    int mt = wave >> 1, nt0 = (wave & 1) * 4;
    for (int k0 = 0; k0 < 64; k0 += 32) {
      union U { v16h v; unsigned u[8]; };
      U ua;
      int arow = mt * 16 + r16;
#pragma unroll
      for (int p = 0; p < 8; ++p)
        ua.u[p] = *(const unsigned*)&sQ[arow * 64 + k0 + frag_k(p, hi)];
#pragma unroll
      for (int a = 0; a < 4; ++a) {
        U ub;
        int bcol = (nt0 + a) * 16 + r16;
#pragma unroll
        for (int p = 0; p < 8; ++p)
          ub.u[p] = *(const unsigned*)&sKV[bcol * 64 + k0 + frag_k(p, hi)];
        acc[a] = __builtin_amdgcn_wmma_f32_16x16x32_f16(
            false, ua.v, false, ub.v, (short)0, acc[a], false, false);
      }
    }
#pragma unroll
    for (int a = 0; a < 4; ++a) {
      int n = (nt0 + a) * 16 + r16;
      int tk = sIdx[n];
#pragma unroll
      for (int p = 0; p < 8; ++p) {
        int m = mt * 16 + p + hi * 8;
        float val = acc[a][p];
        if (sIdx[m] == tk) val = kMASK;   // shared-QK self mask
        sDots[m * 128 + n] = val;
      }
    }
  }
  __syncthreads();

  // rows 0..63: logsumexp + softmax in place. threads 64..191: load V^T over sKV.
  if (tid < 64) {
    int r = tid;
    float mx = -3.4e38f;
    for (int n = 0; n < 128; ++n) mx = fmaxf(mx, sDots[r * 128 + n]);
    float s = 0.f;
    for (int n = 0; n < 128; ++n) {
      float e = expf(sDots[r * 128 + n] - mx);
      s += e; sDots[r * 128 + n] = e;
    }
    float inv = 1.0f / s;
    for (int n = 0; n < 128; ++n) sDots[r * 128 + n] *= inv;
    sLse[r] = mx + logf(s);
  } else if (tid < 192) {
    int jj = tid - 64;
    int ch = (jj < 64) ? c : ((c + kNCHUNK - 1) & (kNCHUNK - 1));
    int i = pRow[ch * 64 + (jj & 63)];
    int t = i & (kT - 1);
    const float* src = vv + ((size_t)(b * kT + t)) * kD + h * kDH;
    for (int e = 0; e < kDH; ++e)
      sKV[e * 128 + jj] = (_Float16)src[e];   // transposed: [e][col]
  }
  __syncthreads();

  // bo = probs @ v : 64x64, K=128. wave w: mtile=w>>1, ntiles=(w&1)*2..+1
  {
    int mt = wave >> 1, nt0 = (wave & 1) * 2;
    v8f acc[2] = {};
    for (int k0 = 0; k0 < 128; k0 += 32) {
      union Uh { v16h v; _Float16 hh[16]; } ua;
      union Uu { v16h v; unsigned u[8]; };
      int arow = mt * 16 + r16;
#pragma unroll
      for (int p = 0; p < 8; ++p) {
        int k = frag_k(p, hi);
        ua.hh[2 * p]     = (_Float16)sDots[arow * 128 + k0 + k];
        ua.hh[2 * p + 1] = (_Float16)sDots[arow * 128 + k0 + k + 1];
      }
#pragma unroll
      for (int a = 0; a < 2; ++a) {
        Uu ub;
        int ncol = (nt0 + a) * 16 + r16;   // = e
#pragma unroll
        for (int p = 0; p < 8; ++p)
          ub.u[p] = *(const unsigned*)&sKV[ncol * 128 + k0 + frag_k(p, hi)];
        acc[a] = __builtin_amdgcn_wmma_f32_16x16x32_f16(
            false, ua.v, false, ub.v, (short)0, acc[a], false, false);
      }
    }
    // scatter straight to unsorted hash slots (fuses the `undo` gather)
#pragma unroll
    for (int a = 0; a < 2; ++a) {
      int n = (nt0 + a) * 16 + r16;
#pragma unroll
      for (int p = 0; p < 8; ++p) {
        int m = mt * 16 + p + hi * 8;
        int i = pRow[c * 64 + m];
        ohash[((size_t)bh * kSORTN + i) * kDH + n] = acc[a][p];
      }
    }
  }
  if (tid < 64) {
    int i = pRow[c * 64 + tid];
    lseo[(size_t)bh * kSORTN + i] = sLse[tid];
  }
}

// ---------------- combine hash rounds (softmax over LSE) -> f16 ----------------
__global__ __launch_bounds__(64) void combine_kernel(
    const float* __restrict__ oh, const float* __restrict__ lse,
    _Float16* __restrict__ attn) {
  int t = blockIdx.x, bh = blockIdx.y;
  int b = bh >> 3, h = bh & 7;
  __shared__ float w[kNHASH];
  if (threadIdx.x == 0) {
    float l[kNHASH], mx = -3.4e38f;
    for (int r = 0; r < kNHASH; ++r) {
      l[r] = lse[(size_t)bh * kSORTN + r * kT + t];
      mx = fmaxf(mx, l[r]);
    }
    float s = 0.f;
    for (int r = 0; r < kNHASH; ++r) { l[r] = expf(l[r] - mx); s += l[r]; }
    for (int r = 0; r < kNHASH; ++r) w[r] = l[r] / s;
  }
  __syncthreads();
  int e = threadIdx.x;
  float acc = 0.f;
  for (int r = 0; r < kNHASH; ++r)
    acc += oh[((size_t)bh * kSORTN + r * kT + t) * kDH + e] * w[r];
  attn[((size_t)(b * kT + t)) * kD + h * kDH + e] = (_Float16)acc;
}

// ---------------- final classifier ----------------
__global__ __launch_bounds__(64) void classifier_kernel(
    const float* __restrict__ x1, const float* __restrict__ x2,
    const float* __restrict__ w, const float* __restrict__ bias,
    float* __restrict__ out) {
  int cc = blockIdx.x, b = blockIdx.y;
  size_t base = (size_t)b * kT * kD;   // token 0 of batch b
  float p = 0.f;
  for (int d = threadIdx.x; d < kD; d += 64)
    p += (x1[base + d] + x2[base + d]) * w[d * kNLABELS + cc];
  __shared__ float red[2];
  for (int o = 16; o > 0; o >>= 1) p += __shfl_down(p, o, 32);
  if ((threadIdx.x & 31) == 0) red[threadIdx.x >> 5] = p;
  __syncthreads();
  if (threadIdx.x == 0) out[b * kNLABELS + cc] = red[0] + red[1] + bias[cc];
}

// ---------------- host launcher ----------------
extern "C" void kernel_launch(void* const* d_in, const int* in_sizes, int n_in,
                              void* d_out, int out_size, void* d_ws, size_t ws_size,
                              hipStream_t stream) {
  const int*   ids   = (const int*)d_in[0];
  const float* tok   = (const float*)d_in[1];
  const float* pos   = (const float*)d_in[2];
  const float* rot   = (const float*)d_in[3];
  const float* Wqk   = (const float*)d_in[4];
  const float* Wv    = (const float*)d_in[5];
  const float* Wout  = (const float*)d_in[6];
  const float* bout  = (const float*)d_in[7];
  const float* lnf_s = (const float*)d_in[8];
  const float* lnf_b = (const float*)d_in[9];
  const float* lng_s = (const float*)d_in[10];
  const float* lng_b = (const float*)d_in[11];
  const float* ffw1  = (const float*)d_in[12];
  const float* ffb1  = (const float*)d_in[13];
  const float* ffw2  = (const float*)d_in[14];
  const float* ffb2  = (const float*)d_in[15];
  const float* clsw  = (const float*)d_in[16];
  const float* clsb  = (const float*)d_in[17];

  constexpr size_t MB = 1u << 20;
  constexpr size_t SZ_ACT  = (size_t)kROWS * kD * sizeof(float);    // 16 MB
  constexpr size_t SZ_ACTH = (size_t)kROWS * kD * sizeof(_Float16); //  8 MB
  char* ws = (char*)d_ws;
  float*     x1     = (float*)(ws);                       // 16 MB
  float*     x2     = (float*)(ws + 16 * MB);             // 16 MB
  float*     qkb    = (float*)(ws + 32 * MB);             // 16 MB
  float*     vvb    = (float*)(ws + 48 * MB);             // 16 MB
  _Float16*  xn16   = (_Float16*)(ws + 64 * MB);          //  8 MB
  _Float16*  attn16 = (_Float16*)(ws + 72 * MB);          //  8 MB
  int*       perm   = (int*)(ws + 80 * MB);               //  1 MB
  float*     lseb   = (float*)(ws + 81 * MB);             //  1 MB
  char*      bigreg = ws + 82 * MB;                       // 64 MB: ohash | ff1(f16)
  float*     ohash  = (float*)bigreg;
  _Float16*  ff16   = (_Float16*)bigreg;
  _Float16*  wqkT   = (_Float16*)(ws + 146 * MB);         //  4 MB (8 x 512KB)
  _Float16*  wvT    = (_Float16*)(ws + 150 * MB);         //  4 MB
  _Float16*  woT    = (_Float16*)(ws + 154 * MB);         //  4 MB
  _Float16*  w1T    = (_Float16*)(ws + 158 * MB);         // 16 MB (8 x 2MB)
  _Float16*  w2T    = (_Float16*)(ws + 174 * MB);         // 16 MB  -> ends 190MB

  // one-time (per launch) weight cast+transpose to f16 [N][K]
  for (int l = 0; l < kDEPTH; ++l) {
    cast_transpose_kernel<<<dim3(kD / 32, kD / 32), 256, 0, stream>>>(
        Wqk + (size_t)l * kD * kD, wqkT + (size_t)l * kD * kD, kD, kD);
    cast_transpose_kernel<<<dim3(kD / 32, kD / 32), 256, 0, stream>>>(
        Wv + (size_t)l * kD * kD, wvT + (size_t)l * kD * kD, kD, kD);
    cast_transpose_kernel<<<dim3(kD / 32, kD / 32), 256, 0, stream>>>(
        Wout + (size_t)l * kD * kD, woT + (size_t)l * kD * kD, kD, kD);
    cast_transpose_kernel<<<dim3(kD / 32, kFF / 32), 256, 0, stream>>>(
        ffw1 + (size_t)l * kD * kFF, w1T + (size_t)l * kD * kFF, kD, kFF);
    cast_transpose_kernel<<<dim3(kFF / 32, kD / 32), 256, 0, stream>>>(
        ffw2 + (size_t)l * kFF * kD, w2T + (size_t)l * kFF * kD, kFF, kD);
  }

  embed_kernel<<<kROWS, 256, 0, stream>>>(ids, tok, pos, x1, x2);

  for (int l = 0; l < kDEPTH; ++l) {
    // y1 = x1 + LSHAttn(LN_f(x2))
    layernorm_kernel<<<kROWS, 256, 0, stream>>>(x2, lnf_s + l * kD, lnf_b + l * kD, xn16);
    gemm_f16wmma<<<dim3(kROWS / 128, kD / 64), 256, 0, stream>>>(
        xn16, wqkT + (size_t)l * kD * kD, nullptr, nullptr, qkb, nullptr,
        kROWS, kD, kD, 0);
    gemm_f16wmma<<<dim3(kROWS / 128, kD / 64), 256, 0, stream>>>(
        xn16, wvT + (size_t)l * kD * kD, nullptr, nullptr, vvb, nullptr,
        kROWS, kD, kD, 0);
    lsh_sort_kernel<<<kBH, 256, 0, stream>>>(
        qkb, rot + (size_t)l * kDH * kNHASH * (kNB / 2), perm);
    lsh_attn_kernel<<<dim3(kNCHUNK, kBH), 256, 0, stream>>>(qkb, vvb, perm, ohash, lseb);
    combine_kernel<<<dim3(kT, kBH), 64, 0, stream>>>(ohash, lseb, attn16);
    gemm_f16wmma<<<dim3(kROWS / 128, kD / 64), 256, 0, stream>>>(
        attn16, woT + (size_t)l * kD * kD, bout + l * kD, /*res=*/x1, /*out=*/x1,
        nullptr, kROWS, kD, kD, 0);

    // y2 = x2 + FFN(LN_g(y1))
    layernorm_kernel<<<kROWS, 256, 0, stream>>>(x1, lng_s + l * kD, lng_b + l * kD, xn16);
    gemm_f16wmma<<<dim3(kROWS / 128, kFF / 64), 256, 0, stream>>>(
        xn16, w1T + (size_t)l * kD * kFF, ffb1 + l * kFF, nullptr,
        nullptr, ff16, kROWS, kFF, kD, /*gelu=*/1);
    gemm_f16wmma<<<dim3(kROWS / 128, kD / 64), 256, 0, stream>>>(
        ff16, w2T + (size_t)l * kFF * kD, ffb2 + l * kD, /*res=*/x2, /*out=*/x2,
        nullptr, kROWS, kD, kFF, 0);
  }

  classifier_kernel<<<dim3(kNLABELS, kB), 64, 0, stream>>>(
      x1, x2, clsw, clsb, (float*)d_out);
}
